// DiffusedAttention_64828236366588
// MI455X (gfx1250) — matrix operand order repaired and checked by reading, not compiled, not run on hardware
//
#include <hip/hip_runtime.h>
#include <hip/hip_bf16.h>

// ---------------------------------------------------------------------------
// DiffusedAttention on MI455X (gfx1250, wave32).
// Dense matmuls: v_wmma_f32_16x16x32_bf16, fp32 accumulate.
//  - K staged in 128-wide chunks; W chunk stored TRANSPOSED in LDS so B
//    fragments are contiguous (2x ds_load_b128 per fragment).
//  - Each wave owns 4 M-tiles (64 rows x 16 cols) -> 16 independent WMMAs
//    per barrier pair, B fragments reused across tiles.
//  - Interior tiles stage via float4 global loads + packed bf16 converts.
// Graph propagation: edge-parallel global_atomic_add_f32 (L2-resident:
// h = 25.6MB << 192MB L2).
// ---------------------------------------------------------------------------

typedef __attribute__((ext_vector_type(16))) __bf16 v16bf;
typedef __attribute__((ext_vector_type(8)))  __bf16 v8bf;
typedef __attribute__((ext_vector_type(4)))  __bf16 v4bf;
typedef __attribute__((ext_vector_type(8)))  float  v8f;

#define HDIM 128
#define HOPS 3      // K+1
#define NHEAD 4
#define HD 32
#define MT 4        // M-tiles per wave (64 rows per block)
#define KCH 128     // K chunk staged per barrier round

static __device__ __forceinline__ v4bf cvt4(float4 v) {
  v4bf r;
  r[0] = (__bf16)v.x; r[1] = (__bf16)v.y;
  r[2] = (__bf16)v.z; r[3] = (__bf16)v.w;
  return r;
}

// ---------------------------------------------------------------------------
// D = act(A[M x K] @ W[K x Nc] + bias)
// A fp32 row-major (row stride lda), W fp32 row-major (K x Nc), D stride ldd.
// grid = (ceil(M/64), ceil(Nc/128), Z); block = 256 (8 waves).
// ---------------------------------------------------------------------------
__global__ __launch_bounds__(256) void gemm_bf16_wmma(
    const float* __restrict__ A, const float* __restrict__ W,
    const float* __restrict__ bias, float* __restrict__ D,
    int M, int K, int Nc, int lda, int ldd,
    long strideAz, long strideWz, long strideBz, long strideDz,
    int act) {
  A    += (long)blockIdx.z * strideAz;
  W    += (long)blockIdx.z * strideWz;
  bias += (long)blockIdx.z * strideBz;
  D    += (long)blockIdx.z * strideDz;

  const int mbase  = blockIdx.x * (16 * MT);
  const int n0base = blockIdx.y * 128;
  const int wave   = threadIdx.x >> 5;
  const int lane   = threadIdx.x & 31;
  const int n0     = n0base + wave * 16;      // wave-uniform

  __shared__ __bf16 sA[16 * MT][KCH];         // 16KB
  __shared__ __bf16 sWt[128][KCH];            // 32KB, TRANSPOSED: [n][k]

  v8f acc[MT];
#pragma unroll
  for (int t = 0; t < MT; ++t) acc[t] = (v8f){};

  const int m   = lane & 15;                  // row within 16-tile
  const int kbA = (lane >> 4) * 8;            // A frag K-half select
  const int ncl = wave * 16 + (lane & 15);    // col within 128 panel
  const int kbB = (lane >> 4) * 8;            // B frag K-half select

  const bool fullM = (mbase + 16 * MT) <= M;
  const bool fullN = (n0base + 128) <= Nc;

  for (int k0 = 0; k0 < K; k0 += KCH) {
    const bool fullK = (k0 + KCH) <= K;

    // ---- stage A chunk (64 x 128) fp32 -> bf16 ----
    if (fullM && fullK) {                     // vectorized fast path
      for (int i = threadIdx.x; i < (16 * MT * KCH) / 4; i += 256) {
        int r = i >> 5;                       // KCH/4 = 32 float4 per row
        int c = (i & 31) << 2;
        float4 v = *(const float4*)(A + (long)(mbase + r) * lda + (k0 + c));
        *(v4bf*)&sA[r][c] = cvt4(v);
      }
    } else {                                  // guarded boundary path
      for (int i = threadIdx.x; i < 16 * MT * KCH; i += 256) {
        int r = i >> 7, c = i & (KCH - 1);
        int gm = mbase + r, gk = k0 + c;
        float v = (gm < M && gk < K) ? A[(long)gm * lda + gk] : 0.f;
        sA[r][c] = (__bf16)v;
      }
    }
    // ---- stage W chunk (128 x 128) fp32 -> bf16, transposed ----
    if (fullN && fullK) {
      for (int i = threadIdx.x; i < (KCH * 128) / 4; i += 256) {
        int r = i >> 5;                       // k
        int c = (i & 31) << 2;                // n (coalesced float4 read)
        float4 v = *(const float4*)(W + (long)(k0 + r) * Nc + (n0base + c));
        v4bf b = cvt4(v);
        sWt[c + 0][r] = b[0]; sWt[c + 1][r] = b[1];
        sWt[c + 2][r] = b[2]; sWt[c + 3][r] = b[3];
      }
    } else {
      for (int i = threadIdx.x; i < KCH * 128; i += 256) {
        int r = i >> 7, c = i & 127;
        int gk = k0 + r, gn = n0base + c;
        float v = (gk < K && gn < Nc) ? W[(long)gk * Nc + gn] : 0.f;
        sWt[c][r] = (__bf16)v;
      }
    }
    __syncthreads();

    if (n0 < Nc) {                            // uniform per wave, EXEC full
#pragma unroll
      for (int kk = 0; kk < KCH; kk += 32) {
        union { v16bf v; v8bf h[2]; } b;
        b.h[0] = *(const v8bf*)&sWt[ncl][kk + kbB];
        b.h[1] = *(const v8bf*)&sWt[ncl][kk + 16 + kbB];
#pragma unroll
        for (int t = 0; t < MT; ++t) {
          union { v16bf v; v8bf h[2]; } a;
          a.h[0] = *(const v8bf*)&sA[t * 16 + m][kk + kbA];
          a.h[1] = *(const v8bf*)&sA[t * 16 + m][kk + 16 + kbA];
          acc[t] = __builtin_amdgcn_wmma_f32_16x16x32_bf16(
              false, a.v, false, b.v, (short)0, acc[t], false, false);
        }
      }
    }
    __syncthreads();
  }

  if (n0 < Nc) {
    const int ncol = n0 + (lane & 15);
    if (ncol < Nc) {
      const float bv = bias[ncol];
#pragma unroll
      for (int t = 0; t < MT; ++t) {
#pragma unroll
        for (int v = 0; v < 8; ++v) {
          int gm = mbase + t * 16 + v + (lane >> 4) * 8;  // C/D: M split at 8
          if (gm < M) {
            float o = acc[t][v] + bv;
            if (act == 1) o = (o > 0.f) ? o : 0.01f * o;  // leaky relu
            D[(long)gm * ldd + ncol] = o;
          }
        }
      }
    }
  }
}

// ---------------------------------------------------------------------------
// Graph-normalization kernels
// ---------------------------------------------------------------------------
__global__ void k_fill(float* p, float v, long n) {
  long i = (long)blockIdx.x * blockDim.x + threadIdx.x;
  if (i < n) p[i] = v;
}
__global__ void k_deg_edges(const int* __restrict__ col, float* deg, int E) {
  int e = blockIdx.x * blockDim.x + threadIdx.x;
  if (e < E) atomicAdd(&deg[col[e]], 1.0f);
}
__global__ void k_dinv(const float* __restrict__ deg, float* dinv, float* dinv2,
                       float* ldeg, int n) {
  int i = blockIdx.x * blockDim.x + threadIdx.x;
  if (i < n) {
    float d  = fmaxf(deg[i], 1e-12f);
    float di = rsqrtf(d);
    dinv[i]  = di;
    dinv2[i] = di * di;
    ldeg[i]  = di * di;            // self-loop contribution to laplacian diag
  }
}
__global__ void k_wn_ldeg(const int* __restrict__ row, const int* __restrict__ col,
                          const float* __restrict__ dinv, float* wn, float* ldeg, int E) {
  int e = blockIdx.x * blockDim.x + threadIdx.x;
  if (e < E) {
    float w = dinv[row[e]] * dinv[col[e]];
    wn[e] = w;
    atomicAdd(&ldeg[row[e]], w);
  }
}

// ---------------------------------------------------------------------------
// prop_L:  out[i] = (ldeg[i]-dinv2[i]) * in[i]  -  sum_{e:col=i} wn[e]*in[row[e]]
// float4 over the 128 channels (32 float4 per node)
// ---------------------------------------------------------------------------
__global__ void k_prop_init(const float* __restrict__ in, float* __restrict__ out,
                            const float* __restrict__ ldeg,
                            const float* __restrict__ dinv2,
                            int n, int ldin, int ldout) {
  long idx = (long)blockIdx.x * blockDim.x + threadIdx.x;   // over n*32
  if (idx < (long)n * 32) {
    int i = (int)(idx >> 5), c = ((int)idx & 31) << 2;
    float s = ldeg[i] - dinv2[i];
    float4 v = *(const float4*)(in + (long)i * ldin + c);
    float4 o = make_float4(s * v.x, s * v.y, s * v.z, s * v.w);
    *(float4*)(out + (long)i * ldout + c) = o;
  }
}
__global__ void k_prop_edges(const int* __restrict__ row, const int* __restrict__ col,
                             const float* __restrict__ wn,
                             const float* __restrict__ in, float* __restrict__ out,
                             int E, int ldin, int ldout) {
  long gid = (long)blockIdx.x * blockDim.x + threadIdx.x;
  int  e   = (int)(gid >> 5);
  int  l   = (int)(gid & 31);        // lane handles 4 channels (float4)
  if (e < E) {
    int   r = row[e], c = col[e];
    float w = wn[e];
    float4 v = *(const float4*)(in + (long)r * ldin + l * 4);
    float* dst = out + (long)c * ldout + l * 4;
    atomicAdd(dst + 0, -w * v.x);
    atomicAdd(dst + 1, -w * v.y);
    atomicAdd(dst + 2, -w * v.z);
    atomicAdd(dst + 3, -w * v.w);
  }
}

// ---------------------------------------------------------------------------
// Global layer-norm (single mean/var over ALL elements), deterministic 2-stage
// red[0]=mean red[1]=inv_std; partials at red+8 (1024 x {sum,sumsq})
// ---------------------------------------------------------------------------
__global__ __launch_bounds__(256) void k_ln_partial(const float* __restrict__ x,
                                                    long n, float* red) {
  __shared__ float ss[256], sq[256];
  const long n4 = n >> 2;
  float s = 0.f, q = 0.f;
  for (long i = (long)blockIdx.x * 256 + threadIdx.x; i < n4; i += 1024L * 256) {
    float4 v = *(const float4*)(x + i * 4);
    s += v.x + v.y + v.z + v.w;
    q += v.x * v.x + v.y * v.y + v.z * v.z + v.w * v.w;
  }
  if (blockIdx.x == 0 && threadIdx.x == 0) {        // deterministic tail
    for (long j = n4 * 4; j < n; ++j) { float v = x[j]; s += v; q += v * v; }
  }
  ss[threadIdx.x] = s; sq[threadIdx.x] = q; __syncthreads();
  for (int st = 128; st > 0; st >>= 1) {
    if (threadIdx.x < st) { ss[threadIdx.x] += ss[threadIdx.x + st];
                            sq[threadIdx.x] += sq[threadIdx.x + st]; }
    __syncthreads();
  }
  if (threadIdx.x == 0) { red[8 + blockIdx.x * 2] = ss[0];
                          red[8 + blockIdx.x * 2 + 1] = sq[0]; }
}
__global__ __launch_bounds__(256) void k_ln_final(float* red, long n) {
  __shared__ float ss[256], sq[256];
  float s = 0.f, q = 0.f;
  for (int i = threadIdx.x; i < 1024; i += 256) {
    s += red[8 + 2 * i]; q += red[8 + 2 * i + 1];
  }
  ss[threadIdx.x] = s; sq[threadIdx.x] = q; __syncthreads();
  for (int st = 128; st > 0; st >>= 1) {
    if (threadIdx.x < st) { ss[threadIdx.x] += ss[threadIdx.x + st];
                            sq[threadIdx.x] += sq[threadIdx.x + st]; }
    __syncthreads();
  }
  if (threadIdx.x == 0) {
    float m = ss[0] / (float)n;
    float v = sq[0] / (float)n - m * m;
    red[0] = m; red[1] = rsqrtf(v + 1e-5f);
  }
}
__global__ void k_ln_apply(float* x, long n, const float* __restrict__ red) {
  long i = (long)blockIdx.x * blockDim.x + threadIdx.x;   // over n/4
  const long n4 = n >> 2;
  if (i < n4) {
    float mean = red[0], inv = red[1];
    float4 v = *(const float4*)(x + i * 4);
    v.x = (v.x - mean) * inv; v.y = (v.y - mean) * inv;
    v.z = (v.z - mean) * inv; v.w = (v.w - mean) * inv;
    *(float4*)(x + i * 4) = v;
  }
  if (i == 0) {
    float mean = red[0], inv = red[1];
    for (long j = n4 * 4; j < n; ++j) x[j] = (x[j] - mean) * inv;
  }
}

// ---------------------------------------------------------------------------
// Per-node hop attention (3x3 per head), one wave per node, lane = HD channel.
// out = attn + tok (residual fused). Safe to alias out with q: each wave
// reads its node's q before writing.
// ---------------------------------------------------------------------------
__global__ __launch_bounds__(256) void k_attention(
    const float* __restrict__ tok, const float* __restrict__ q,
    const float* __restrict__ k,   const float* __restrict__ Bp,
    const float* __restrict__ hb,  float* __restrict__ outres, int n) {
  int node = (int)(((long)blockIdx.x * blockDim.x + threadIdx.x) >> 5);
  int lane = threadIdx.x & 31;
  if (node >= n) return;
  const long base = (long)node * (HOPS * HDIM);
  const float invs = 0.17677669529663687f;   // 1/sqrt(32)
#pragma unroll
  for (int h = 0; h < NHEAD; ++h) {
    float qa[HOPS], kb[HOPS], tb[HOPS], bias[HOPS];
#pragma unroll
    for (int a = 0; a < HOPS; ++a) {
      long off = base + a * HDIM + h * HD + lane;
      qa[a] = q[off]; kb[a] = k[off]; tb[a] = tok[off];
      bias[a] = Bp[a] * hb[h * HOPS + a];
    }
    float sc[HOPS][HOPS];
#pragma unroll
    for (int a = 0; a < HOPS; ++a)
#pragma unroll
      for (int b = 0; b < HOPS; ++b) {
        float d = qa[a] * kb[b];
#pragma unroll
        for (int s = 16; s > 0; s >>= 1) d += __shfl_xor(d, s, 32);
        sc[a][b] = tanhf(d * invs);
      }
#pragma unroll
    for (int a = 0; a < HOPS; ++a) {
      float acc = tb[a];                      // residual
#pragma unroll
      for (int b = 0; b < HOPS; ++b) acc += sc[a][b] * tb[b] * bias[b];
      outres[base + a * HDIM + h * HD + lane] = acc;
    }
  }
}

// ---------------------------------------------------------------------------
// Misc small kernels (float4 over channels)
// ---------------------------------------------------------------------------
__global__ void k_copy_h_to_hop0(const float* __restrict__ h, float* __restrict__ tok, int n) {
  long idx = (long)blockIdx.x * blockDim.x + threadIdx.x;   // over n*32
  if (idx < (long)n * 32) {
    int i = (int)(idx >> 5), c = ((int)idx & 31) << 2;
    *(float4*)(tok + (long)i * (HOPS * HDIM) + c) = *(const float4*)(h + (long)i * HDIM + c);
  }
}
__global__ void k_sum_hops(const float* __restrict__ tok, float* __restrict__ h, int n) {
  long idx = (long)blockIdx.x * blockDim.x + threadIdx.x;   // over n*32
  if (idx < (long)n * 32) {
    int i = (int)(idx >> 5), c = ((int)idx & 31) << 2;
    const float* p = tok + (long)i * (HOPS * HDIM) + c;
    float4 a = *(const float4*)(p);
    float4 b = *(const float4*)(p + HDIM);
    float4 d = *(const float4*)(p + 2 * HDIM);
    float4 o = make_float4(a.x + b.x + d.x, a.y + b.y + d.y,
                           a.z + b.z + d.z, a.w + b.w + d.w);
    *(float4*)(h + (long)i * HDIM + c) = o;
  }
}
__global__ __launch_bounds__(256) void k_logsoftmax(
    const float* __restrict__ logits, float* __restrict__ out, int n, int C) {
  int r = (int)(((long)blockIdx.x * blockDim.x + threadIdx.x) >> 5);
  int lane = threadIdx.x & 31;
  if (r >= n) return;
  const float* src = logits + (long)r * C;
  float mx = -3.4e38f;
  for (int c = lane; c < C; c += 32) mx = fmaxf(mx, src[c]);
  for (int s = 16; s > 0; s >>= 1) mx = fmaxf(mx, __shfl_xor(mx, s, 32));
  float se = 0.f;
  for (int c = lane; c < C; c += 32) se += expf(src[c] - mx);
  for (int s = 16; s > 0; s >>= 1) se += __shfl_xor(se, s, 32);
  float lse = mx + logf(se);
  for (int c = lane; c < C; c += 32) out[(long)r * C + c] = src[c] - lse;
}

// ---------------------------------------------------------------------------
// Host orchestration
// ---------------------------------------------------------------------------
static inline void run_gemm(const float* A, const float* W, const float* bias,
                            float* D, int M, int K, int Nc, int lda, int ldd,
                            long sAz, long sWz, long sBz, long sDz, int Z,
                            int act, hipStream_t s) {
  dim3 g((M + 16 * MT - 1) / (16 * MT), (Nc + 127) / 128, Z);
  gemm_bf16_wmma<<<g, 256, 0, s>>>(A, W, bias, D, M, K, Nc, lda, ldd,
                                   sAz, sWz, sBz, sDz, act);
}
static inline void run_ln(float* buf, long total, float* red, hipStream_t s) {
  k_ln_partial<<<1024, 256, 0, s>>>(buf, total, red);
  k_ln_final<<<1, 256, 0, s>>>(red, total);
  long blocks = ((total >> 2) + 255) / 256;
  k_ln_apply<<<(unsigned)blocks, 256, 0, s>>>(buf, total, red);
}

extern "C" void kernel_launch(void* const* d_in, const int* in_sizes, int n_in,
                              void* d_out, int out_size, void* d_ws, size_t ws_size,
                              hipStream_t stream) {
  const float* x    = (const float*)d_in[0];
  const int*   ei   = (const int*)d_in[1];
  const float* encW = (const float*)d_in[2];
  const float* encB = (const float*)d_in[3];
  const float* decW = (const float*)d_in[4];
  const float* decB = (const float*)d_in[5];

  struct Layer {
    const float *linW1, *linb1, *linW2, *linb2, *Wq, *bq, *Wk, *bk,
                *f1W, *f1b, *f2W, *f2b, *B, *hb;
  } L[4];
  {
    int b = 6;
    for (int l = 0; l < 4; ++l) {
      L[l].linW1 = (const float*)d_in[b + 0];  L[l].linb1 = (const float*)d_in[b + 1];
      L[l].linW2 = (const float*)d_in[b + 2];  L[l].linb2 = (const float*)d_in[b + 3];
      L[l].Wq    = (const float*)d_in[b + 4];  L[l].bq    = (const float*)d_in[b + 5];
      L[l].Wk    = (const float*)d_in[b + 6];  L[l].bk    = (const float*)d_in[b + 7];
      L[l].f1W   = (const float*)d_in[b + 8];  L[l].f1b   = (const float*)d_in[b + 9];
      L[l].f2W   = (const float*)d_in[b + 10]; L[l].f2b   = (const float*)d_in[b + 11];
      L[l].B     = (const float*)d_in[b + 12]; L[l].hb    = (const float*)d_in[b + 13];
      b += 14;
    }
  }

  const int H = HDIM, C = 40;
  const int F_IN = in_sizes[2] / H;          // enc_W is (F_IN, H)
  const int N    = in_sizes[0] / F_IN;
  const int E    = in_sizes[1] / 2;
  const int* row = ei;
  const int* col = ei + E;

  // ---- workspace carve (256B aligned slabs) ----
  char* wp = (char*)d_ws;
  auto carve = [&](size_t bytes) -> float* {
    float* p = (float*)wp;
    wp += (bytes + 255) & ~(size_t)255;
    return p;
  };
  float* red   = carve((8 + 2048) * sizeof(float));       // mean/inv + partials
  float* wn    = carve((size_t)E * sizeof(float));
  float* dinv  = carve((size_t)N * sizeof(float));        // also deg during setup
  float* dinv2 = carve((size_t)N * sizeof(float));
  float* ldeg  = carve((size_t)N * sizeof(float));
  float* hbuf  = carve((size_t)N * H * sizeof(float));
  float* tokA  = carve((size_t)N * HOPS * H * sizeof(float));
  float* tokB  = carve((size_t)N * HOPS * H * sizeof(float));
  float* tokC  = carve((size_t)N * HOPS * H * sizeof(float));
  const int CH = 37504;                                    // FFN row chunk (mult of 64)
  float* ffn   = carve((size_t)CH * 4 * H * sizeof(float));
  float* logit = carve((size_t)N * C * sizeof(float));
  (void)ws_size; (void)n_in;

  const long NH4  = (long)N * 32;            // N*H/4 float4 work-items
  const int  T4_NH  = (int)((NH4 + 255) / 256);
  const int  T256_E = (E + 255) / 256;
  const int  T256_N = (N + 255) / 256;
  const int  EDGE_BLK = (int)(((long)E * 32 + 255) / 256);
  const long TOK_elems = (long)N * HOPS * H;
  const long NH_elems  = (long)N * H;

  // ---- graph normalization: deg -> dinv -> wn, ldeg ----
  k_fill<<<T256_N, 256, 0, stream>>>(dinv, 1.0f, N);              // deg (self loop)
  k_deg_edges<<<T256_E, 256, 0, stream>>>(col, dinv, E);
  k_dinv<<<T256_N, 256, 0, stream>>>(dinv, dinv, dinv2, ldeg, N); // in-place deg->dinv
  k_wn_ldeg<<<T256_E, 256, 0, stream>>>(row, col, dinv, wn, ldeg, E);

  // ---- encoder: h = x @ enc_W + enc_b ----
  run_gemm(x, encW, encB, hbuf, N, F_IN, H, F_IN, H, 0, 0, 0, 0, 1, 0, stream);

  const int LDT = HOPS * H;   // 384
  for (int l = 0; l < 4; ++l) {
    // hop tokens: hop0 = h; hop1 = L@hop0; hop2 = L@hop1
    k_copy_h_to_hop0<<<T4_NH, 256, 0, stream>>>(hbuf, tokA, N);
    for (int kk = 1; kk < HOPS; ++kk) {
      const float* in  = tokA + (kk - 1) * H;
      float*       out = tokA + kk * H;
      k_prop_init<<<T4_NH, 256, 0, stream>>>(in, out, ldeg, dinv2, N, LDT, LDT);
      k_prop_edges<<<EDGE_BLK, 256, 0, stream>>>(row, col, wn, in, out, E, LDT, LDT);
    }
    // global LN over all (N,3,H) elements
    run_ln(tokA, TOK_elems, red, stream);

    // per-hop 2-layer MLP (Z=3 batches the hop dimension)
    run_gemm(tokA, L[l].linW1, L[l].linb1, tokB, N, H, H, LDT, LDT,
             H, (long)H * H, H, H, HOPS, /*lrelu*/1, stream);
    run_gemm(tokB, L[l].linW2, L[l].linb2, tokA, N, H, H, LDT, LDT,
             H, (long)H * H, H, H, HOPS, 0, stream);

    // Q / K projections over flattened (3N, H)
    run_gemm(tokA, L[l].Wq, L[l].bq, tokB, HOPS * N, H, H, H, H,
             0, 0, 0, 0, 1, 0, stream);
    run_gemm(tokA, L[l].Wk, L[l].bk, tokC, HOPS * N, H, H, H, H,
             0, 0, 0, 0, 1, 0, stream);

    // per-node attention + residual -> tokB, then global LN
    k_attention<<<(N * 32 + 255) / 256, 256, 0, stream>>>(
        tokA, tokB, tokC, L[l].B, L[l].hb, tokB, N);
    run_ln(tokB, TOK_elems, red, stream);

    // FFN 128 -> 512 -> 128 (row-chunked to bound the 4H intermediate)
    const int MR = HOPS * N;
    for (int r0 = 0; r0 < MR; r0 += CH) {
      int m = (MR - r0 < CH) ? (MR - r0) : CH;
      run_gemm(tokB + (long)r0 * H, L[l].f1W, L[l].f1b, ffn,
               m, H, 4 * H, H, 4 * H, 0, 0, 0, 0, 1, /*lrelu*/1, stream);
      run_gemm(ffn, L[l].f2W, L[l].f2b, tokC + (long)r0 * H,
               m, 4 * H, H, 4 * H, H, 0, 0, 0, 0, 1, 0, stream);
    }
    // h = sum over hop axis
    k_sum_hops<<<T4_NH, 256, 0, stream>>>(tokC, hbuf, N);
  }

  // final global LN, decoder, log-softmax
  run_ln(hbuf, NH_elems, red, stream);
  run_gemm(hbuf, decW, decB, logit, N, H, C, H, C, 0, 0, 0, 0, 1, 0, stream);
  k_logsoftmax<<<(N * 32 + 255) / 256, 256, 0, stream>>>(logit, (float*)d_out, N, C);
}
